// GNN_Detector_60473139527896
// MI455X (gfx1250) — matrix operand order repaired and checked by reading, not compiled, not run on hardware
//
#include <hip/hip_runtime.h>
#include <hip/hip_bf16.h>
#include <cstdint>

// ---- problem constants ----
#define NB   8
#define NN   2048
#define FF   128
#define CC   32
#define KK   512
#define C1O  16
#define C2W  5
#define C2O  32
#define DENSE 128
#define SPLIT 4
#define KSEG (NN / SPLIT)   // 512

typedef __attribute__((ext_vector_type(16))) __bf16 v16bf;
typedef __attribute__((ext_vector_type(8)))  float  v8f;

__device__ __forceinline__ uint16_t f2bf(float f) {
  union { float f; uint32_t u; } cv; cv.f = f;
  uint32_t u = cv.u;
  uint32_t r = u + 0x7FFFu + ((u >> 16) & 1u);   // round-to-nearest-even
  return (uint16_t)(r >> 16);
}

// ---------------- 1) A_tilde f32 -> bf16 (read once, reuse 4x; fits in 192MB L2) ----------------
__global__ void __launch_bounds__(256) k_cvt_bf16(const float* __restrict__ A,
                                                  uint16_t* __restrict__ Abf) {
  size_t id   = (size_t)blockIdx.x * blockDim.x + threadIdx.x;
  size_t base = id * 4;                       // 33,554,432 elems / 4 per thread
  float4 v = *(const float4*)(A + base);
  uint64_t p = (uint64_t)f2bf(v.x) | ((uint64_t)f2bf(v.y) << 16) |
               ((uint64_t)f2bf(v.z) << 32) | ((uint64_t)f2bf(v.w) << 48);
  *(uint64_t*)(Abf + base) = p;
}

// ---------------- 2) HWt[b][c][n] = (H @ W)^T in bf16 ----------------
__global__ void __launch_bounds__(128) k_hwt(const float* __restrict__ H, int ldH, int Cin,
                                             const float* __restrict__ W,
                                             uint16_t* __restrict__ HWt) {
  __shared__ float Ws[FF * CC];
  int b = blockIdx.y;
  int n = blockIdx.x * blockDim.x + threadIdx.x;
  for (int idx = threadIdx.x; idx < Cin * CC; idx += blockDim.x) Ws[idx] = W[idx];
  __syncthreads();
  const float* hrow = H + ((size_t)b * NN + n) * ldH;
  float acc[CC];
#pragma unroll
  for (int c = 0; c < CC; ++c) acc[c] = 0.f;
  for (int f = 0; f < Cin; ++f) {
    float hv = hrow[f];
#pragma unroll
    for (int c = 0; c < CC; ++c) acc[c] += hv * Ws[f * CC + c];
  }
  for (int c = 0; c < CC; ++c)
    HWt[((size_t)b * CC + c) * NN + n] = f2bf(acc[c]);
}

// ---------------- 3) WMMA GEMM (K-split x4): Zpart[seg] = A[:,kseg] @ HW[kseg,:] ----------------
// one wave = one 16-row strip x 32 cols x one K-segment; 4096 waves (8b x 128 tiles x 4 segs)
// seg/b are derived from blockIdx.x so loop bounds are SGPR-uniform (scalar loop, EXEC
// statically all-ones around every WMMA as the ISA requires).
__global__ void __launch_bounds__(256) k_gcn_gemm(const uint16_t* __restrict__ Abf,
                                                  const uint16_t* __restrict__ HWt,
                                                  float* __restrict__ Zpart) {
  const int blk = blockIdx.x;                 // 0..511
  const int seg = blk >> 7;                   // 0..3   (scalar)
  const int b   = (blk >> 4) & 7;             // 0..7   (scalar)
  const int wiw = threadIdx.x >> 5;           // wave in workgroup, 0..7
  const int i0  = ((((blk & 15) << 3) | wiw) << 4);  // row tile * 16
  const int lane = threadIdx.x & 31;
  const int m   = lane & 15;
  const int hi  = (lane >> 4) & 1;

  const uint16_t* arow  = Abf + ((size_t)b << 22) + (size_t)(i0 + m) * NN;
  const int aksel = hi ? 8 : 0;                      // ISA 16-bit A 16x32 layout
  const uint16_t* brow0 = HWt + ((size_t)b * CC + m)      * NN;   // n = lane%16, cols 0..15
  const uint16_t* brow1 = HWt + ((size_t)b * CC + 16 + m) * NN;   // cols 16..31
  const int bksel = hi ? 8 : 0;                      // B 32x16: same K order as A (0..7/16..23, +8 hi)

  v8f acc0 = {};
  v8f acc1 = {};
  union Frag { v16bf v; uint32_t u[8]; };

  const int kbeg = seg * KSEG;
  const int kend = kbeg + KSEG;
  for (int k0 = kbeg; k0 < kend; k0 += 32) {
    Frag a, b0, b1;
#pragma unroll
    for (int v = 0; v < 8; ++v) {
      int ak = k0 + aksel + ((v < 4) ? (2 * v) : (8 + 2 * v)); // pairs K: 0..7 / 16..23 (+8 hi)
      a.u[v]  = *(const uint32_t*)(arow + ak);
      int bk = k0 + bksel + ((v < 4) ? (2 * v) : (8 + 2 * v));
      b0.u[v] = *(const uint32_t*)(brow0 + bk);
      b1.u[v] = *(const uint32_t*)(brow1 + bk);
    }
    __builtin_prefetch(arow + k0 + 64, 0, 1);   // global_prefetch_b8 on next A block
    acc0 = __builtin_amdgcn_wmma_f32_16x16x32_bf16(false, a.v, false, b0.v,
                                                   (short)0, acc0, false, false);
    acc1 = __builtin_amdgcn_wmma_f32_16x16x32_bf16(false, a.v, false, b1.v,
                                                   (short)0, acc1, false, false);
  }

#pragma unroll
  for (int r = 0; r < 8; ++r) {
    int i = i0 + r + (hi ? 8 : 0);                 // C/D layout: vgpr r, hi half -> M+8
    float* zp = Zpart + (((size_t)seg * NB + b) * NN + i) * CC;
    zp[m]      = acc0[r];
    zp[16 + m] = acc1[r];
  }
}

// ---------------- 3b) combine K-split partials, fuse Dinv scale + tanh ----------------
__global__ void __launch_bounds__(256) k_combine(const float* __restrict__ Zpart,
                                                 const float* __restrict__ Dinv,
                                                 float* __restrict__ gco, int layer) {
  int id = blockIdx.x * blockDim.x + threadIdx.x;   // NB*NN*CC = 524288
  int c = id & (CC - 1);
  int i = (id >> 5) & (NN - 1);
  int b = id >> 16;
  float v = 0.f;
#pragma unroll
  for (int s = 0; s < SPLIT; ++s)
    v += Zpart[(((size_t)s * NB + b) * NN + i) * CC + c];
  float d = Dinv[(size_t)b * NN + i];
  gco[((size_t)b * NN + i) * (4 * CC) + layer * CC + c] = tanhf(d * v);
}

// ---------------- 4) stable descending top-K gather ----------------
__global__ void __launch_bounds__(1024) k_topk(const float* __restrict__ gco,
                                               float* __restrict__ topk) {
  __shared__ float s[NN];
  int b = blockIdx.x, tid = threadIdx.x;
  for (int j = tid; j < NN; j += 1024) s[j] = gco[((size_t)b * NN + j) * (4 * CC) + 127];
  __syncthreads();
  for (int i = tid; i < NN; i += 1024) {
    float si = s[i];
    int rank = 0;
    for (int j = 0; j < NN; ++j) {
      float sj = s[j];
      rank += (sj > si) || (sj == si && j < i);    // stable tie-break like lax.top_k
    }
    if (rank < KK) {
      const float4* src = (const float4*)(gco + ((size_t)b * NN + i) * (4 * CC));
      float4* dst = (float4*)(topk + ((size_t)b * KK + rank) * (4 * CC));
      for (int c = 0; c < CC; ++c) dst[c] = src[c];
    }
  }
}

// ---------------- 5) pointwise conv1 + relu + pair maxpool ----------------
__global__ void __launch_bounds__(256) k_conv1pool(const float* __restrict__ topk,
                                                   const float* __restrict__ k1,
                                                   const float* __restrict__ b1,
                                                   float* __restrict__ p) {
  int id = blockIdx.x * blockDim.x + threadIdx.x;   // 8*256*16 = 32768
  int o  = id & (C1O - 1);
  int kk = (id >> 4) & 255;
  int b  = id >> 12;
  float mx = 0.f;
#pragma unroll
  for (int h = 0; h < 2; ++h) {
    int k = 2 * kk + h;
    const float* x = topk + ((size_t)b * KK + k) * (4 * CC);
    float acc = b1[o];
    for (int d = 0; d < 4 * CC; ++d) acc += x[d] * k1[d * C1O + o];
    acc = fmaxf(acc, 0.f);
    mx = fmaxf(mx, acc);
  }
  p[((size_t)b * 256 + kk) * C1O + o] = mx;
}

// ---------------- 6) width-5 SAME conv 16->32 + relu ----------------
__global__ void __launch_bounds__(256) k_conv2(const float* __restrict__ p,
                                               const float* __restrict__ k2,
                                               const float* __restrict__ b2,
                                               float* __restrict__ c2) {
  int id = blockIdx.x * blockDim.x + threadIdx.x;   // 8*256*32 = 65536
  int o = id & (C2O - 1);
  int t = (id >> 5) & 255;
  int b = id >> 13;
  float acc = b2[o];
  for (int w = 0; w < C2W; ++w) {
    int tt = t + w - 2;
    if (tt >= 0 && tt < 256) {
      const float* pr = p + ((size_t)b * 256 + tt) * C1O;
      for (int i = 0; i < C1O; ++i) acc += pr[i] * k2[(w * C1O + i) * C2O + o];
    }
  }
  c2[((size_t)b * 256 + t) * C2O + o] = fmaxf(acc, 0.f);
}

// ---------------- 7) dense (8192 -> 128) + relu, 8-way split-i LDS reduction ----------------
__global__ void __launch_bounds__(1024) k_dense(const float* __restrict__ c2,
                                                const float* __restrict__ dW,
                                                const float* __restrict__ db,
                                                float* __restrict__ h) {
  __shared__ float red[8][DENSE];
  int b = blockIdx.x;
  int j = threadIdx.x & (DENSE - 1);
  int seg = threadIdx.x >> 7;                       // 8 segments of 1024 i's
  const float* cf = c2 + (size_t)b * (256 * C2O);
  float acc = 0.f;
  int ibeg = seg * 1024, iend = ibeg + 1024;
  for (int i = ibeg; i < iend; ++i) acc += cf[i] * dW[(size_t)i * DENSE + j];
  red[seg][j] = acc;
  __syncthreads();
  if (seg == 0) {
    float t = db[j];
#pragma unroll
    for (int s = 0; s < 8; ++s) t += red[s][j];
    h[b * DENSE + j] = fmaxf(t, 0.f);
  }
}

// ---------------- 8) logits, softmax, argmax ----------------
__global__ void k_head(const float* __restrict__ h, const float* __restrict__ oW,
                       const float* __restrict__ ob, float* __restrict__ out) {
  int b = threadIdx.x;
  if (b < NB) {
    float l0 = ob[0], l1 = ob[1];
    const float* hr = h + b * DENSE;
    for (int j = 0; j < DENSE; ++j) { l0 += hr[j] * oW[j * 2]; l1 += hr[j] * oW[j * 2 + 1]; }
    float mx = fmaxf(l0, l1);
    float e0 = expf(l0 - mx), e1 = expf(l1 - mx), s = e0 + e1;
    out[b * 2]          = e0 / s;       // pos_score
    out[b * 2 + 1]      = e1 / s;
    out[16 + b * 2]     = l0;           // logits
    out[16 + b * 2 + 1] = l1;
    out[32 + b]         = (l0 >= l1) ? 0.0f : 1.0f;  // argmax (first-index ties)
  }
}

extern "C" void kernel_launch(void* const* d_in, const int* in_sizes, int n_in,
                              void* d_out, int out_size, void* d_ws, size_t ws_size,
                              hipStream_t stream) {
  const float* Dinv = (const float*)d_in[0];
  const float* A    = (const float*)d_in[1];
  const float* X    = (const float*)d_in[2];
  const float* W1   = (const float*)d_in[3];
  const float* W2   = (const float*)d_in[4];
  const float* W3   = (const float*)d_in[5];
  const float* W4   = (const float*)d_in[6];
  const float* k1   = (const float*)d_in[7];
  const float* b1   = (const float*)d_in[8];
  const float* k2   = (const float*)d_in[9];
  const float* b2   = (const float*)d_in[10];
  const float* dW   = (const float*)d_in[11];
  const float* db   = (const float*)d_in[12];
  const float* oW   = (const float*)d_in[13];
  const float* ob   = (const float*)d_in[14];

  char* ws = (char*)d_ws;
  uint16_t* Abf   = (uint16_t*)(ws);                    // 67,108,864 B
  uint16_t* HWt   = (uint16_t*)(ws + 67108864);         //  1,048,576 B
  float*    gco   = (float*)(ws + 68157440);            //  8,388,608 B
  float*    topk  = (float*)(ws + 76546048);            //  2,097,152 B
  float*    pbuf  = (float*)(ws + 78643200);            //    131,072 B
  float*    c2b   = (float*)(ws + 78774272);            //    262,144 B
  float*    hbuf  = (float*)(ws + 79036416);            //      4,096 B
  float*    Zpart = (float*)(ws + 79040512);            //  8,388,608 B  (~83.4 MB total)

  k_cvt_bf16<<<32768, 256, 0, stream>>>(A, Abf);

  const float* Hs[4]  = {X, gco, gco + 32, gco + 64};   // layer l reads cols (l-1)*32
  const float* Wl[4]  = {W1, W2, W3, W4};
  int          Cin[4] = {FF, CC, CC, CC};
  for (int l = 0; l < 4; ++l) {
    k_hwt<<<dim3(NN / 128, NB), 128, 0, stream>>>(Hs[l], 4 * CC, Cin[l], Wl[l], HWt);
    k_gcn_gemm<<<512, 256, 0, stream>>>(Abf, HWt, Zpart);
    k_combine<<<(NB * NN * CC) / 256, 256, 0, stream>>>(Zpart, Dinv, gco, l);
  }

  k_topk<<<NB, 1024, 0, stream>>>(gco, topk);
  k_conv1pool<<<128, 256, 0, stream>>>(topk, k1, b1, pbuf);
  k_conv2<<<256, 256, 0, stream>>>(pbuf, k2, b2, c2b);
  k_dense<<<NB, 1024, 0, stream>>>(c2b, dW, db, hbuf);
  k_head<<<1, 32, 0, stream>>>(hbuf, oW, ob, (float*)d_out);
}
